// Attention_45320494907482
// MI455X (gfx1250) — matrix operand order repaired and checked by reading, not compiled, not run on hardware
//
#include <hip/hip_runtime.h>

typedef __bf16 bf16;
typedef __attribute__((ext_vector_type(16))) __bf16 v16bf;
typedef __attribute__((ext_vector_type(8)))  __bf16 v8bf;
typedef __attribute__((ext_vector_type(4)))  __bf16 v4bf;
typedef __attribute__((ext_vector_type(4)))  float  v4f;
typedef __attribute__((ext_vector_type(8)))  float  v8f;

#define HEADS 12
#define HDIM  64
#define SEQ   1024
#define CDIM  768
#define ATT_SCALE 0.125f   // 64^-0.5

__device__ __forceinline__ v8f wmma_bf16(v16bf a, v16bf b, v8f c) {
  // D(f32 16x16) = A(16x32 bf16) * B(32x16 bf16) + C
  return __builtin_amdgcn_wmma_f32_16x16x32_bf16(false, a, false, b, (short)0, c, false, false);
}

// A-operand fragment of a 16x32 bf16 tile, source row-major with leading dim ld.
// lane: row = lane&15 ; element i holds k = (lane>>4)*16 + i (contiguous)
__device__ __forceinline__ v16bf load_a_frag(const bf16* base, int ld, int row, int k0, int lane) {
  const int half = lane >> 4;
  const bf16* p = base + (size_t)row * ld + k0 + half * 16;
  return *(const v16bf*)(p);
}

// B-operand fragment: B[k][n] = stored[n][k]. Caller passes pointer to stored row n
// (n = lane&15 chosen by caller). element i holds k = k0 + (lane>>4)*16 + i (contiguous).
__device__ __forceinline__ v16bf load_b_frag(const bf16* rowp, int k0, int lane) {
  return *(const v16bf*)(rowp + k0 + ((lane >> 4) << 4));
}

// ---------------------------------------------------------------------------
// Kernel 0: fp32 -> bf16 pre-pack (vectorized, grid-stride)
// ---------------------------------------------------------------------------
__global__ __launch_bounds__(256) void pack_bf16(const float* __restrict__ in,
                                                 bf16* __restrict__ out, int n4) {
  for (int i = blockIdx.x * 256 + threadIdx.x; i < n4; i += gridDim.x * 256) {
    v4f f = *(const v4f*)(in + (size_t)i * 4);
    v4bf o;
    o[0] = (bf16)f[0]; o[1] = (bf16)f[1]; o[2] = (bf16)f[2]; o[3] = (bf16)f[3];
    *(v4bf*)(out + (size_t)i * 4) = o;
  }
}

// ---------------------------------------------------------------------------
// Kernel 1: qkv = x @ w_qkv.T  (bf16 in/out), v stored transposed [b,h,d,n]
// grid(256, 36), block 32 (one wave computes a 16x64 output tile)
// ---------------------------------------------------------------------------
__global__ __launch_bounds__(32) void qkv_gemm(const bf16* __restrict__ xb,
                                               const bf16* __restrict__ wqkv,
                                               bf16* __restrict__ qo,
                                               bf16* __restrict__ ko,
                                               bf16* __restrict__ vto) {
  const int lane = threadIdx.x & 31;
  const int half = lane >> 4, l15 = lane & 15;
  const int m0 = blockIdx.x * 16;
  const int j0 = blockIdx.y * 64;
  v8f z = {};
  v8f acc[4];
  #pragma unroll
  for (int dt = 0; dt < 4; ++dt) acc[dt] = z;

  const bf16* wrow[4];
  #pragma unroll
  for (int dt = 0; dt < 4; ++dt)
    wrow[dt] = wqkv + (size_t)(j0 + dt * 16 + l15) * CDIM;

  for (int kk = 0; kk < CDIM; kk += 32) {
    v16bf a = load_a_frag(xb, CDIM, m0 + l15, kk, lane);
    v16bf bb[4];
    #pragma unroll
    for (int dt = 0; dt < 4; ++dt) bb[dt] = load_b_frag(wrow[dt], kk, lane);
    #pragma unroll
    for (int dt = 0; dt < 4; ++dt) acc[dt] = wmma_bf16(a, bb[dt], acc[dt]);
  }
  #pragma unroll
  for (int dt = 0; dt < 4; ++dt) {
    const int j = j0 + dt * 16 + l15;
    const int t = j / CDIM;            // 0=q, 1=k, 2=v
    const int h = (j % CDIM) / HDIM;
    const int d = j % HDIM;
    #pragma unroll
    for (int r = 0; r < 8; ++r) {
      const int rowg = m0 + r + 8 * half;
      const int b = rowg >> 10, n = rowg & (SEQ - 1);
      const bf16 v = (bf16)acc[dt][r];
      const size_t hb = (size_t)(b * HEADS + h);
      if (t == 0)      qo[(hb * SEQ + n) * HDIM + d] = v;
      else if (t == 1) ko[(hb * SEQ + n) * HDIM + d] = v;
      else             vto[(hb * HDIM + d) * SEQ + n] = v;   // transposed
    }
  }
}

// ---------------------------------------------------------------------------
// Kernel 2: fused talking-heads attention.
// Workgroup = (batch, 16-row tile), 4 waves.
// WMMA stages (QK^T, P@V): wave w owns heads 3w..3w+2.
// Head-mix stages: wave w owns rows 4w..4w+3 (all 12 heads) so every
// S/P element is read from LDS exactly once per workgroup.
// Two L2-resident streaming phases:
//   phase 1: online-softmax stats (running max + rescaled running sum),
//   phase 2: exact P, post-mix, P@V accumulate.
// ---------------------------------------------------------------------------
__global__ __launch_bounds__(128) void attn_core(const bf16* __restrict__ qg,
                                                 const bf16* __restrict__ kg,
                                                 const bf16* __restrict__ vtg,
                                                 const float* __restrict__ mask,
                                                 const float* __restrict__ w_pre,
                                                 const float* __restrict__ w_post,
                                                 bf16* __restrict__ attO) {
  __shared__ float S_lds[HEADS][16][36];   // raw logits (padded rows: bank spread)
  __shared__ bf16  P_lds[HEADS][16][36];   // normalized probs
  __shared__ bf16  T_lds[HEADS][16][32];   // post-mixed probs (WMMA A staging)
  __shared__ float M_lds[HEADS][16];
  __shared__ float Li_lds[HEADS][16];
  __shared__ float wpre_s[HEADS][HEADS];
  __shared__ float wpost_s[HEADS][HEADS];
  __shared__ float pm_s[HEADS];            // row sums of w_pre (mask coefficient)

  const int tid  = threadIdx.x;
  const int lane = tid & 31, wid = tid >> 5;
  const int half = lane >> 4, l15 = lane & 15;
  const int b  = blockIdx.y;
  const int n0 = blockIdx.x * 16;
  const int h0 = wid * 3;                  // heads owned in WMMA stages
  const int mrow = wid * 4 + (lane >> 3);  // row owned in mix stages
  const int c0 = lane & 7;                 // column phase in mix stages

  for (int i = tid; i < HEADS * HEADS; i += 128) {
    ((float*)wpre_s)[i]  = w_pre[i];
    ((float*)wpost_s)[i] = w_post[i];
  }
  __syncthreads();
  if (tid < HEADS) {
    float s = 0.f;
    for (int h = 0; h < HEADS; ++h) s += wpre_s[tid][h];
    pm_s[tid] = s;
  }

  // q fragments stay resident in VGPRs for both phases
  v16bf qf[3][2];
  #pragma unroll
  for (int hh = 0; hh < 3; ++hh) {
    const bf16* qbase = qg + (size_t)(b * HEADS + h0 + hh) * SEQ * HDIM;
    #pragma unroll
    for (int ks = 0; ks < 2; ++ks)
      qf[hh][ks] = load_a_frag(qbase, HDIM, n0 + l15, ks * 32, lane);
  }

  const float* maskrow = mask + ((size_t)b * SEQ + n0 + mrow) * SEQ;
  v8f zf = {};

  // ------------------- phase 1: online softmax statistics -------------------
  float statM[HEADS], statL[HEADS];
  #pragma unroll
  for (int g = 0; g < HEADS; ++g) { statM[g] = -3.0e38f; statL[g] = 0.f; }

  for (int mb = 0; mb < SEQ / 32; ++mb) {
    const int m0 = mb * 32;
    __syncthreads();
    #pragma unroll
    for (int hh = 0; hh < 3; ++hh) {
      const int h = h0 + hh;
      const bf16* kbase = kg + (size_t)(b * HEADS + h) * SEQ * HDIM;
      #pragma unroll
      for (int st = 0; st < 2; ++st) {
        const bf16* krow = kbase + (size_t)(m0 + st * 16 + l15) * HDIM;
        v16bf kb0 = load_b_frag(krow, 0, lane);
        v16bf kb1 = load_b_frag(krow, 32, lane);
        v8f s = zf;
        s = wmma_bf16(qf[hh][0], kb0, s);
        s = wmma_bf16(qf[hh][1], kb1, s);
        #pragma unroll
        for (int r = 0; r < 8; ++r)
          S_lds[h][r + 8 * half][st * 16 + l15] = s[r];
      }
    }
    __syncthreads();
    #pragma unroll
    for (int j = 0; j < 4; ++j) {
      const int col = c0 + 8 * j;
      float sh[HEADS];
      #pragma unroll
      for (int h = 0; h < HEADS; ++h) sh[h] = S_lds[h][mrow][col];
      const float mval = maskrow[m0 + col];
      #pragma unroll
      for (int g = 0; g < HEADS; ++g) {
        float accv = 0.f;
        #pragma unroll
        for (int h = 0; h < HEADS; ++h) accv += wpre_s[g][h] * sh[h];
        const float sp = accv * ATT_SCALE + pm_s[g] * mval;
        const float mn = fmaxf(statM[g], sp);
        statL[g] = statL[g] * __expf(statM[g] - mn) + __expf(sp - mn);
        statM[g] = mn;
      }
    }
  }
  // merge (m,l) across the 8 lanes sharing (g, mrow)
  #pragma unroll
  for (int g = 0; g < HEADS; ++g) {
    float m = statM[g], l = statL[g];
    #pragma unroll
    for (int off = 1; off < 8; off <<= 1) {
      const float mo = __shfl_xor(m, off, 32);
      const float lo = __shfl_xor(l, off, 32);
      const float mn = fmaxf(m, mo);
      l = l * __expf(m - mn) + lo * __expf(mo - mn);
      m = mn;
    }
    if (c0 == 0) { M_lds[g][mrow] = m; Li_lds[g][mrow] = 1.f / l; }
  }
  __syncthreads();

  // ------------------- phase 2: P, post-mix, P@V -------------------
  float Mreg[HEADS], Lireg[HEADS];
  #pragma unroll
  for (int g = 0; g < HEADS; ++g) {
    Mreg[g]  = M_lds[g][mrow];
    Lireg[g] = Li_lds[g][mrow];
  }
  v8f outacc[3][4];
  #pragma unroll
  for (int gg = 0; gg < 3; ++gg)
    #pragma unroll
    for (int dt = 0; dt < 4; ++dt) outacc[gg][dt] = zf;

  for (int mb = 0; mb < SEQ / 32; ++mb) {
    const int m0 = mb * 32;
    __syncthreads();
    #pragma unroll
    for (int hh = 0; hh < 3; ++hh) {
      const int h = h0 + hh;
      const bf16* kbase = kg + (size_t)(b * HEADS + h) * SEQ * HDIM;
      #pragma unroll
      for (int st = 0; st < 2; ++st) {
        const bf16* krow = kbase + (size_t)(m0 + st * 16 + l15) * HDIM;
        v16bf kb0 = load_b_frag(krow, 0, lane);
        v16bf kb1 = load_b_frag(krow, 32, lane);
        v8f s = zf;
        s = wmma_bf16(qf[hh][0], kb0, s);
        s = wmma_bf16(qf[hh][1], kb1, s);
        #pragma unroll
        for (int r = 0; r < 8; ++r)
          S_lds[h][r + 8 * half][st * 16 + l15] = s[r];
      }
    }
    __syncthreads();
    // pre-mix -> exact normalized P
    #pragma unroll
    for (int j = 0; j < 4; ++j) {
      const int col = c0 + 8 * j;
      float sh[HEADS];
      #pragma unroll
      for (int h = 0; h < HEADS; ++h) sh[h] = S_lds[h][mrow][col];
      const float mval = maskrow[m0 + col];
      #pragma unroll
      for (int g = 0; g < HEADS; ++g) {
        float accv = 0.f;
        #pragma unroll
        for (int h = 0; h < HEADS; ++h) accv += wpre_s[g][h] * sh[h];
        const float sp = accv * ATT_SCALE + pm_s[g] * mval;
        P_lds[g][mrow][col] = (bf16)(__expf(sp - Mreg[g]) * Lireg[g]);
      }
    }
    __syncthreads();
    // post-mix -> T (bf16, WMMA A staging)
    #pragma unroll
    for (int j = 0; j < 4; ++j) {
      const int col = c0 + 8 * j;
      float ph[HEADS];
      #pragma unroll
      for (int h = 0; h < HEADS; ++h) ph[h] = (float)P_lds[h][mrow][col];
      #pragma unroll
      for (int g = 0; g < HEADS; ++g) {
        float t = 0.f;
        #pragma unroll
        for (int h = 0; h < HEADS; ++h) t += wpost_s[g][h] * ph[h];
        T_lds[g][mrow][col] = (bf16)t;
      }
    }
    __syncthreads();
    // out[g2] += T[g2] @ v[g2]   (v pre-transposed -> contiguous B frags)
    #pragma unroll
    for (int gg = 0; gg < 3; ++gg) {
      const int g2 = h0 + gg;
      v16bf a = load_a_frag(&T_lds[g2][0][0], 32, l15, 0, lane);
      const bf16* vb = vtg + (size_t)(b * HEADS + g2) * HDIM * SEQ;
      v16bf vbfr[4];
      #pragma unroll
      for (int dt = 0; dt < 4; ++dt)
        vbfr[dt] = load_b_frag(vb + (size_t)(dt * 16 + l15) * SEQ + m0, 0, lane);
      #pragma unroll
      for (int dt = 0; dt < 4; ++dt)
        outacc[gg][dt] = wmma_bf16(a, vbfr[dt], outacc[gg][dt]);
    }
  }

  // store attention output (bf16, [B*N, C] row-major)
  #pragma unroll
  for (int gg = 0; gg < 3; ++gg) {
    const int g2 = h0 + gg;
    #pragma unroll
    for (int dt = 0; dt < 4; ++dt) {
      const int col = g2 * HDIM + dt * 16 + l15;
      #pragma unroll
      for (int r = 0; r < 8; ++r) {
        const int row = n0 + r + 8 * half;
        attO[(size_t)(b * SEQ + row) * CDIM + col] = (bf16)outacc[gg][dt][r];
      }
    }
  }
}

// ---------------------------------------------------------------------------
// Kernel 3: out = attO @ w_proj.T + b_proj  (bf16 A and B, fp32 out)
// grid(256, 12), block 32
// ---------------------------------------------------------------------------
__global__ __launch_bounds__(32) void proj_gemm(const bf16* __restrict__ ao,
                                                const bf16* __restrict__ wp,
                                                const float* __restrict__ bias,
                                                float* __restrict__ out) {
  const int lane = threadIdx.x & 31;
  const int half = lane >> 4, l15 = lane & 15;
  const int m0 = blockIdx.x * 16;
  const int j0 = blockIdx.y * 64;
  v8f z = {};
  v8f acc[4];
  #pragma unroll
  for (int dt = 0; dt < 4; ++dt) acc[dt] = z;

  const bf16* wrow[4];
  #pragma unroll
  for (int dt = 0; dt < 4; ++dt)
    wrow[dt] = wp + (size_t)(j0 + dt * 16 + l15) * CDIM;

  for (int kk = 0; kk < CDIM; kk += 32) {
    v16bf a = load_a_frag(ao, CDIM, m0 + l15, kk, lane);
    v16bf bb[4];
    #pragma unroll
    for (int dt = 0; dt < 4; ++dt) bb[dt] = load_b_frag(wrow[dt], kk, lane);
    #pragma unroll
    for (int dt = 0; dt < 4; ++dt) acc[dt] = wmma_bf16(a, bb[dt], acc[dt]);
  }
  #pragma unroll
  for (int dt = 0; dt < 4; ++dt) {
    const int col = j0 + dt * 16 + l15;
    const float bv = bias[col];
    #pragma unroll
    for (int r = 0; r < 8; ++r) {
      const int rowg = m0 + r + 8 * half;
      out[(size_t)rowg * CDIM + col] = acc[dt][r] + bv;
    }
  }
}

extern "C" void kernel_launch(void* const* d_in, const int* in_sizes, int n_in,
                              void* d_out, int out_size, void* d_ws, size_t ws_size,
                              hipStream_t stream) {
  const float* x      = (const float*)d_in[0];
  const float* mask   = (const float*)d_in[1];
  const float* w_qkv  = (const float*)d_in[2];
  const float* w_proj = (const float*)d_in[3];
  const float* b_proj = (const float*)d_in[4];
  const float* w_pre  = (const float*)d_in[5];
  const float* w_post = (const float*)d_in[6];
  float* out = (float*)d_out;

  const size_t per     = (size_t)4 * HEADS * SEQ * HDIM;   // 3,145,728 elems
  const size_t n_x     = (size_t)4 * SEQ * CDIM;           // == per
  const size_t n_wqkv  = (size_t)3 * CDIM * CDIM;          // 1,769,472
  const size_t n_wproj = (size_t)CDIM * CDIM;              //   589,824

  bf16* q      = (bf16*)d_ws;
  bf16* k      = q + per;
  bf16* vt     = k + per;
  bf16* attO   = vt + per;      // 4096*768 == per
  bf16* xb     = attO + per;
  bf16* wqkvb  = xb + n_x;
  bf16* wprojb = wqkvb + n_wqkv;

  pack_bf16<<<dim3(512), 256, 0, stream>>>(x,      xb,     (int)(n_x     / 4));
  pack_bf16<<<dim3(512), 256, 0, stream>>>(w_qkv,  wqkvb,  (int)(n_wqkv  / 4));
  pack_bf16<<<dim3(256), 256, 0, stream>>>(w_proj, wprojb, (int)(n_wproj / 4));

  qkv_gemm<<<dim3(256, 36), 32, 0, stream>>>(xb, wqkvb, q, k, vt);
  attn_core<<<dim3(SEQ / 16, 4), 128, 0, stream>>>(q, k, vt, mask, w_pre, w_post, attO);
  proj_gemm<<<dim3(256, 12), 32, 0, stream>>>(attO, wprojb, b_proj, out);
}
